// AFTLayer_74208444940329
// MI455X (gfx1250) — matrix operand order repaired and checked by reading, not compiled, not run on hardware
//
#include <hip/hip_runtime.h>
#include <hip/hip_bf16.h>

typedef _Float16 v16h __attribute__((ext_vector_type(16)));
typedef _Float16 h8v  __attribute__((ext_vector_type(8)));
typedef float    v8f  __attribute__((ext_vector_type(8)));

#define N_TOK 784
#define DIM   128
#define FF    512
#define KPAD  800   // 784 padded up to multiple of 32, pad region zeroed

enum { EPI_F32 = 0, EPI_EXP_SPLIT = 1, EPI_BIAS_RES = 2, EPI_BIAS_GELU_SPLIT = 3 };

__device__ __forceinline__ float wave_sum32(float v) {
#pragma unroll
    for (int off = 16; off > 0; off >>= 1) v += __shfl_xor(v, off, 32);
    return v;
}

__device__ __forceinline__ void split_store(float v, _Float16* __restrict__ h,
                                            _Float16* __restrict__ l, size_t idx) {
    _Float16 hv = (_Float16)v;
    h[idx] = hv;
    l[idx] = (_Float16)(v - (float)hv);
}

// ---------------- elementwise kernels ----------------

// x: [C][N] -> t: [N][C]
__global__ void k_transpose_in(const float* __restrict__ x, float* __restrict__ t) {
    int idx = blockIdx.x * blockDim.x + threadIdx.x;
    if (idx >= N_TOK * DIM) return;
    int n = idx / DIM, c = idx - n * DIM;
    t[idx] = x[c * N_TOK + n];
}

// t: [N][C] -> y: [C][N]
__global__ void k_transpose_out(const float* __restrict__ t, float* __restrict__ y) {
    int idx = blockIdx.x * blockDim.x + threadIdx.x;
    if (idx >= N_TOK * DIM) return;
    int c = idx / N_TOK, n = idx - c * N_TOK;
    y[idx] = t[n * DIM + c];
}

// generic f32 -> (hi,lo) f16 split
__global__ void k_split(const float* __restrict__ src, _Float16* __restrict__ h,
                        _Float16* __restrict__ l, int n) {
    int idx = blockIdx.x * blockDim.x + threadIdx.x;
    if (idx >= n) return;
    split_store(src[idx], h, l, idx);
}

__global__ void k_zero_h(_Float16* __restrict__ p, int n) {
    int idx = blockIdx.x * blockDim.x + threadIdx.x;
    if (idx < n) p[idx] = (_Float16)0.0f;
}

// one wave per row of 128 channels; writes split (hi,lo) output
__global__ void k_layernorm(const float* __restrict__ src, const float* __restrict__ g,
                            const float* __restrict__ b, _Float16* __restrict__ dh,
                            _Float16* __restrict__ dl) {
    int n = blockIdx.x;
    int lane = threadIdx.x;
    const float* row = src + n * DIM;
    float v[4];
#pragma unroll
    for (int i = 0; i < 4; ++i) v[i] = row[lane + 32 * i];
    float s = wave_sum32(v[0] + v[1] + v[2] + v[3]);
    float mu = s * (1.0f / DIM);
    float q = 0.f;
#pragma unroll
    for (int i = 0; i < 4; ++i) { float d = v[i] - mu; q += d * d; }
    q = wave_sum32(q);
    float rs = rsqrtf(q * (1.0f / DIM) + 1e-5f);
#pragma unroll
    for (int i = 0; i < 4; ++i) {
        int c = lane + 32 * i;
        split_store((v[i] - mu) * rs * g[c] + b[c], dh, dl, (size_t)n * DIM + c);
    }
}

// qlin <- sigmoid(qlin) (in place, f32)
// ekT/evvT <- split(exp(k)), split(exp(k)*v), stored transposed [D][KPAD]
__global__ void k_qkv_ew(float* __restrict__ qlin, const float* __restrict__ klin,
                         const float* __restrict__ vlin,
                         _Float16* __restrict__ ekTh, _Float16* __restrict__ ekTl,
                         _Float16* __restrict__ evvTh, _Float16* __restrict__ evvTl) {
    int idx = blockIdx.x * blockDim.x + threadIdx.x;
    if (idx >= N_TOK * DIM) return;
    int n = idx / DIM, d = idx - n * DIM;
    float e = expf(klin[idx]);
    size_t tidx = (size_t)d * KPAD + n;
    split_store(e, ekTh, ekTl, tidx);
    split_store(e * vlin[idx], evvTh, evvTl, tidx);
    qlin[idx] = 1.0f / (1.0f + expf(-qlin[idx]));
}

// g <- split(q * num / den)
__global__ void k_gate(const float* __restrict__ q, const float* __restrict__ num,
                       const float* __restrict__ den, _Float16* __restrict__ gh,
                       _Float16* __restrict__ gl) {
    int idx = blockIdx.x * blockDim.x + threadIdx.x;
    if (idx >= N_TOK * DIM) return;
    split_store(q[idx] * num[idx] / den[idx], gh, gl, idx);
}

// ---------------- WMMA GEMM ----------------
// C[M x N] = A[M x K] * B^T, A:[M][K] (hi,lo f16), B:[N][K] (hi,lo f16), K % 32 == 0.
// One wave per 16x16 tile. Split-f16: acc += Ah*Bh + Al*Bh + Ah*Bl (f32 accumulate).
// Fragment layouts per CDNA5 ISA 7.12.2:
//   A 16x32: lane&15 = row M; ka = (h>>3)*16 + (lane>>4)*8 + (h&7)  -> two runs of 8 halfs
//   B 32x16: lane&15 = col N; kb = (h>>3)*16 + (lane>>4)*8 + (h&7)  -> two runs of 8 halfs
//   C/D v8f: row = i + 8*(lane>>4), col = lane&15
template <int EPI>
__global__ void k_gemm(const _Float16* __restrict__ Ah, const _Float16* __restrict__ Al, int lda,
                       const _Float16* __restrict__ Bh, const _Float16* __restrict__ Bl, int ldb,
                       const float* __restrict__ bias, const float* __restrict__ resid,
                       float* __restrict__ Cf, _Float16* __restrict__ Ch,
                       _Float16* __restrict__ Cl, int ldc, int K) {
    const int lane = threadIdx.x & 31;
    const int r    = lane & 15;
    const int hi   = lane >> 4;
    const int nt   = blockIdx.x;
    const int mt   = blockIdx.y;

    const _Float16* arh = Ah + (size_t)(mt * 16 + r) * lda + hi * 8;
    const _Float16* arl = Al + (size_t)(mt * 16 + r) * lda + hi * 8;
    const _Float16* brh = Bh + (size_t)(nt * 16 + r) * ldb + hi * 8;
    const _Float16* brl = Bl + (size_t)(nt * 16 + r) * ldb + hi * 8;

    v8f acc = {};
    for (int kk = 0; kk < K; kk += 32) {
        h8v a0 = *(const h8v*)(arh + kk);
        h8v a1 = *(const h8v*)(arh + kk + 16);
        h8v a2 = *(const h8v*)(arl + kk);
        h8v a3 = *(const h8v*)(arl + kk + 16);
        h8v b0 = *(const h8v*)(brh + kk);
        h8v b1 = *(const h8v*)(brh + kk + 16);
        h8v b2 = *(const h8v*)(brl + kk);
        h8v b3 = *(const h8v*)(brl + kk + 16);
        v16h avh = __builtin_shufflevector(a0, a1, 0,1,2,3,4,5,6,7,8,9,10,11,12,13,14,15);
        v16h avl = __builtin_shufflevector(a2, a3, 0,1,2,3,4,5,6,7,8,9,10,11,12,13,14,15);
        v16h bvh = __builtin_shufflevector(b0, b1, 0,1,2,3,4,5,6,7,8,9,10,11,12,13,14,15);
        v16h bvl = __builtin_shufflevector(b2, b3, 0,1,2,3,4,5,6,7,8,9,10,11,12,13,14,15);
        acc = __builtin_amdgcn_wmma_f32_16x16x32_f16(false, avh, false, bvh, (short)0, acc, false, false);
        acc = __builtin_amdgcn_wmma_f32_16x16x32_f16(false, avl, false, bvh, (short)0, acc, false, false);
        acc = __builtin_amdgcn_wmma_f32_16x16x32_f16(false, avh, false, bvl, (short)0, acc, false, false);
    }

#pragma unroll
    for (int i = 0; i < 8; ++i) {
        size_t row = (size_t)(mt * 16 + i + hi * 8);
        int col = nt * 16 + r;
        float v = acc[i];
        if (EPI == EPI_F32) {
            Cf[row * ldc + col] = v;
        } else if (EPI == EPI_EXP_SPLIT) {
            split_store(expf(v), Ch, Cl, row * ldc + col);
        } else if (EPI == EPI_BIAS_RES) {
            Cf[row * ldc + col] = v + bias[col] + resid[row * ldc + col];
        } else if (EPI == EPI_BIAS_GELU_SPLIT) {
            v += bias[col];
            v = 0.5f * v * (1.0f + erff(v * 0.70710678118654752f));
            split_store(v, Ch, Cl, row * ldc + col);
        }
    }
}

// ---------------- host side ----------------

extern "C" void kernel_launch(void* const* d_in, const int* in_sizes, int n_in,
                              void* d_out, int out_size, void* d_ws, size_t ws_size,
                              hipStream_t stream) {
    (void)in_sizes; (void)n_in; (void)out_size; (void)ws_size;

    const float* x    = (const float*)d_in[0];
    const float* wq   = (const float*)d_in[1];
    const float* wk   = (const float*)d_in[2];
    const float* wv   = (const float*)d_in[3];
    const float* wo   = (const float*)d_in[4];
    const float* bo   = (const float*)d_in[5];
    const float* u    = (const float*)d_in[6];
    const float* v    = (const float*)d_in[7];
    const float* ln1g = (const float*)d_in[8];
    const float* ln1b = (const float*)d_in[9];
    const float* ln2g = (const float*)d_in[10];
    const float* ln2b = (const float*)d_in[11];
    const float* w1   = (const float*)d_in[12];
    const float* b1   = (const float*)d_in[13];
    const float* w2   = (const float*)d_in[14];
    const float* b2   = (const float*)d_in[15];
    float* out = (float*)d_out;

    const int NC = N_TOK * DIM;  // 100352

    char* cur = (char*)d_ws;
    auto alloc_f = [&](size_t n) { float* p = (float*)cur; cur += n * sizeof(float); return p; };
    auto alloc_h = [&](size_t n) { _Float16* p = (_Float16*)cur; cur += n * sizeof(_Float16); return p; };

    // f32 buffers
    float* t    = alloc_f(NC);
    float* qlin = alloc_f(NC);
    float* klin = alloc_f(NC);
    float* vlin = alloc_f(NC);
    float* num  = alloc_f(NC);
    float* den  = alloc_f(NC);
    float* t2   = alloc_f(NC);
    float* t3   = t;  // t dead after residual into t2

    // f16 (hi,lo) buffers — all sizes multiples of 16 elems keep 16B/32B alignment
    _Float16* xnh = alloc_h(NC);     _Float16* xnl = alloc_h(NC);      // also reused as xn2
    _Float16* wqh = alloc_h(DIM*DIM); _Float16* wql = alloc_h(DIM*DIM);
    _Float16* wkh = alloc_h(DIM*DIM); _Float16* wkl = alloc_h(DIM*DIM);
    _Float16* wvh = alloc_h(DIM*DIM); _Float16* wvl = alloc_h(DIM*DIM);
    _Float16* woh = alloc_h(DIM*DIM); _Float16* wol = alloc_h(DIM*DIM);
    _Float16* uh  = alloc_h(NC);     _Float16* ul  = alloc_h(NC);
    _Float16* vh  = alloc_h(NC);     _Float16* vl  = alloc_h(NC);
    _Float16* w1h = alloc_h(FF*DIM); _Float16* w1l = alloc_h(FF*DIM);
    _Float16* w2h = alloc_h(DIM*FF); _Float16* w2l = alloc_h(DIM*FF);
    _Float16* gh  = alloc_h(NC);     _Float16* gl  = alloc_h(NC);
    _Float16* hffh = alloc_h((size_t)N_TOK*FF);
    _Float16* hffl = alloc_h((size_t)N_TOK*FF);
    // contiguous: P pair (zeroed together), then ekT/evvT group (zeroed together)
    _Float16* Ph   = alloc_h((size_t)N_TOK*KPAD);
    _Float16* Pl   = alloc_h((size_t)N_TOK*KPAD);
    _Float16* ekTh = alloc_h((size_t)DIM*KPAD);
    _Float16* ekTl = alloc_h((size_t)DIM*KPAD);
    _Float16* evvTh= alloc_h((size_t)DIM*KPAD);
    _Float16* evvTl= alloc_h((size_t)DIM*KPAD);

    dim3 b256(256);
    dim3 gNC((NC + 255) / 256);
    dim3 wave(32);
    auto grid_ew = [](int n) { return dim3((n + 255) / 256); };

    // input transpose + weight splits (fresh every call; deterministic)
    k_transpose_in<<<gNC, b256, 0, stream>>>(x, t);
    k_split<<<grid_ew(DIM*DIM), b256, 0, stream>>>(wq, wqh, wql, DIM*DIM);
    k_split<<<grid_ew(DIM*DIM), b256, 0, stream>>>(wk, wkh, wkl, DIM*DIM);
    k_split<<<grid_ew(DIM*DIM), b256, 0, stream>>>(wv, wvh, wvl, DIM*DIM);
    k_split<<<grid_ew(DIM*DIM), b256, 0, stream>>>(wo, woh, wol, DIM*DIM);
    k_split<<<grid_ew(NC), b256, 0, stream>>>(u, uh, ul, NC);
    k_split<<<grid_ew(NC), b256, 0, stream>>>(v, vh, vl, NC);
    k_split<<<grid_ew(FF*DIM), b256, 0, stream>>>(w1, w1h, w1l, FF*DIM);
    k_split<<<grid_ew(FF*DIM), b256, 0, stream>>>(w2, w2h, w2l, FF*DIM);
    // zero K-padded buffers (pad columns must be 0; bulk-zero then overwrite)
    {
        int nP = 2 * N_TOK * KPAD;          // Ph+Pl contiguous
        int nT = 4 * DIM * KPAD;            // ekT/evvT hi+lo contiguous
        k_zero_h<<<grid_ew(nP), b256, 0, stream>>>(Ph, nP);
        k_zero_h<<<grid_ew(nT), b256, 0, stream>>>(ekTh, nT);
    }

    k_layernorm<<<N_TOK, 32, 0, stream>>>(t, ln1g, ln1b, xnh, xnl);

    // q/k/v projections: [784x128] = xn * W^T, K=128
    k_gemm<EPI_F32><<<dim3(DIM/16, N_TOK/16), wave, 0, stream>>>(
        xnh, xnl, DIM, wqh, wql, DIM, nullptr, nullptr, qlin, nullptr, nullptr, DIM, DIM);
    k_gemm<EPI_F32><<<dim3(DIM/16, N_TOK/16), wave, 0, stream>>>(
        xnh, xnl, DIM, wkh, wkl, DIM, nullptr, nullptr, klin, nullptr, nullptr, DIM, DIM);
    k_gemm<EPI_F32><<<dim3(DIM/16, N_TOK/16), wave, 0, stream>>>(
        xnh, xnl, DIM, wvh, wvl, DIM, nullptr, nullptr, vlin, nullptr, nullptr, DIM, DIM);

    k_qkv_ew<<<gNC, b256, 0, stream>>>(qlin, klin, vlin, ekTh, ekTl, evvTh, evvTl);

    // P = exp(u * v^T)  [784 x 784], K=128, split output with row stride KPAD
    k_gemm<EPI_EXP_SPLIT><<<dim3(N_TOK/16, N_TOK/16), wave, 0, stream>>>(
        uh, ul, DIM, vh, vl, DIM, nullptr, nullptr, nullptr, Ph, Pl, KPAD, DIM);

    // num = P * evv, den = P * ek  [784 x 128], K=KPAD (pad contributes 0*0)
    k_gemm<EPI_F32><<<dim3(DIM/16, N_TOK/16), wave, 0, stream>>>(
        Ph, Pl, KPAD, evvTh, evvTl, KPAD, nullptr, nullptr, num, nullptr, nullptr, DIM, KPAD);
    k_gemm<EPI_F32><<<dim3(DIM/16, N_TOK/16), wave, 0, stream>>>(
        Ph, Pl, KPAD, ekTh, ekTl, KPAD, nullptr, nullptr, den, nullptr, nullptr, DIM, KPAD);

    k_gate<<<gNC, b256, 0, stream>>>(qlin, num, den, gh, gl);

    // t2 = t + g * wo^T + bo
    k_gemm<EPI_BIAS_RES><<<dim3(DIM/16, N_TOK/16), wave, 0, stream>>>(
        gh, gl, DIM, woh, wol, DIM, bo, t, t2, nullptr, nullptr, DIM, DIM);

    k_layernorm<<<N_TOK, 32, 0, stream>>>(t2, ln2g, ln2b, xnh, xnl);  // xn2 reuses xn

    // hff = gelu(xn2 * w1^T + b1)  [784 x 512], split output
    k_gemm<EPI_BIAS_GELU_SPLIT><<<dim3(FF/16, N_TOK/16), wave, 0, stream>>>(
        xnh, xnl, DIM, w1h, w1l, DIM, b1, nullptr, nullptr, hffh, hffl, FF, DIM);

    // t3 = t2 + hff * w2^T + b2  [784 x 128], K=512
    k_gemm<EPI_BIAS_RES><<<dim3(DIM/16, N_TOK/16), wave, 0, stream>>>(
        hffh, hffl, FF, w2h, w2l, FF, b2, t2, t3, nullptr, nullptr, DIM, FF);

    k_transpose_out<<<gNC, b256, 0, stream>>>(t3, out);
}
